// inception_block_dcn_v4_58815282152119
// MI455X (gfx1250) — matrix operand order repaired and checked by reading, not compiled, not run on hardware
//
#include <hip/hip_runtime.h>
#include <hip/hip_bf16.h>

// ---------------------------------------------------------------------------
// MI455X / gfx1250 implementation.
// All tensor contractions run on v_wmma_f32_16x16x32_f16 (f16 in, f32 acc).
// BN+ReLU fused into every consumer's operand load; BN stats are separate
// bandwidth-bound reduction kernels (float4 loads).
// Operand path: LDS tiles laid out channel-contiguous so every WMMA fragment
// is fed by 16-byte ds_load_b128s; weights staged to LDS once per block in
// GEMM-K order. Deformable conv does data-dependent fp32 bilinear gathers
// (validity+sigmoid mask folded into corner weights) feeding the 9*Cin GEMM.
// ---------------------------------------------------------------------------

typedef _Float16 half_t;
typedef __attribute__((ext_vector_type(16))) _Float16 v16h;
typedef __attribute__((ext_vector_type(8)))  _Float16 v8h;
typedef __attribute__((ext_vector_type(8)))  float    v8f;

#define XT      112   // x-positions per block (2 blocks per 224-wide row)
#define BLOCK_T 224   // 7 wave32s: 7 N-subtiles of 16
#define PADH    8     // LDS row padding (halves): 16B, keeps b128 alignment

static __device__ __forceinline__ int iclamp(int v, int lo, int hi) {
    return v < lo ? lo : (v > hi ? hi : v);
}

static __device__ __forceinline__ v16h cat8(v8h lo, v8h hi) {
    return __builtin_shufflevector(lo, hi, 0, 1, 2, 3, 4, 5, 6, 7,
                                           8, 9, 10, 11, 12, 13, 14, 15);
}

// ---------------------------------------------------------------------------
// Per-channel mean / rsqrt(var+eps) over (N, H, W). One block per channel.
// Pure bandwidth: float4 loads, LDS tree reduction.
// ---------------------------------------------------------------------------
__global__ __launch_bounds__(256) void channel_stats(
    const float* __restrict__ t, int C, int HWn, int Bn,
    float* __restrict__ st /* [2C]: mean then rstd */)
{
    __shared__ float s1[256], s2[256];
    const int c = blockIdx.x, tid = threadIdx.x;
    const int q = HWn >> 2;                    // HWn divisible by 4
    float sum = 0.f, sq = 0.f;
    for (int n = 0; n < Bn; ++n) {
        const float4* p = (const float4*)(t + ((size_t)n * C + c) * (size_t)HWn);
        for (int i = tid; i < q; i += 256) {
            float4 v = p[i];
            sum += v.x + v.y + v.z + v.w;
            sq  += v.x * v.x + v.y * v.y + v.z * v.z + v.w * v.w;
        }
    }
    s1[tid] = sum; s2[tid] = sq;
    __syncthreads();
    for (int off = 128; off > 0; off >>= 1) {
        if (tid < off) { s1[tid] += s1[tid + off]; s2[tid] += s2[tid + off]; }
        __syncthreads();
    }
    if (tid == 0) {
        float N   = (float)(Bn * HWn);
        float m   = s1[0] / N;
        float var = s2[0] / N - m * m;
        st[c]     = m;
        st[C + c] = rsqrtf(var + 1e-5f);
    }
}

// ---------------------------------------------------------------------------
// Conv-as-GEMM with fused BN+ReLU on the input operand.
//   out[m, (b,y,x)] = sum_{tap,c} W[m,c,tap] * relu(bn(in[c, y+ty-pY, x+tx-pX]))
// K ordering: K = tap*Cin + c, chunks of 32 per WMMA.
// LDS: act tile [ty][x][c] (channel-contiguous, padded rows) and weights
// [m][K] (padded rows, zero-filled for m >= Cout) -> all fragments are
// 16B ds_load_b128s.
// ---------------------------------------------------------------------------
template<int TY, int TX>
__global__ __launch_bounds__(BLOCK_T) void bnrelu_conv_wmma(
    const float* __restrict__ in,  const float* __restrict__ st,
    const float* __restrict__ gamma, const float* __restrict__ beta,
    const float* __restrict__ wt,  const float* __restrict__ bias,
    float* __restrict__ out,
    int Cin, int cinShift, int Cout, int coutTotal, int coutOffset,
    int padY, int padX, int H, int W)
{
    extern __shared__ __align__(16) char smem_raw[];
    const int LW    = XT + TX - 1;
    const int CinP  = Cin + PADH;
    const int Ktot  = Cin * TY * TX;
    const int KtotP = Ktot + PADH;
    half_t* ldsA = (half_t*)smem_raw;             // [TY*LW][CinP]
    half_t* ldsW = ldsA + TY * LW * CinP;         // [32][KtotP]

    const int b  = blockIdx.z, y = blockIdx.y, x0 = blockIdx.x * XT;
    const int tid = threadIdx.x, lane = tid & 31, wid = tid >> 5;
    const float* mean = st;
    const float* rstd = st + Cin;

    // -------- stage weights into [m][K] f16 (zero rows for m >= Cout) -----
    for (int idx = tid; idx < 32 * Ktot; idx += BLOCK_T) {
        int m = idx / Ktot;
        int K = idx - m * Ktot;
        int tap = K >> cinShift;
        int c   = K & (Cin - 1);
        ldsW[m * KtotP + K] =
            (m < Cout) ? (half_t)wt[(m * Cin + c) * (TY * TX) + tap]
                       : (half_t)0.f;
    }

    // -------- stage BN+ReLU'd halo tile: read coalesced over x, write [x][c]
    const int total = TY * Cin * LW;
    for (int idx = tid; idx < total; idx += BLOCK_T) {
        int ty  = idx / (Cin * LW);
        int rem = idx - ty * (Cin * LW);
        int c   = rem / LW;
        int xi  = rem - c * LW;
        int gy  = y + ty - padY;
        int gx  = x0 + xi - padX;
        float v = 0.f;
        if (gy >= 0 && gy < H && gx >= 0 && gx < W) {
            float r = in[(((size_t)b * Cin + c) * H + gy) * (size_t)W + gx];
            r = (r - mean[c]) * rstd[c] * gamma[c] + beta[c];
            v = fmaxf(r, 0.f);
        }
        ldsA[(ty * LW + xi) * CinP + c] = (half_t)v;
    }
    __syncthreads();

    const int nChunk = Ktot >> 5;
    const int laneHi = (lane >= 16) ? 1 : 0;
    const int nIdx   = lane & 15;
    const int xi0    = wid * 16 + nIdx;   // position inside tile
    const int mRow   = lane & 15;
    const half_t* wRow0 = ldsW + mRow * KtotP;
    const half_t* wRow1 = ldsW + (mRow + 16) * KtotP;

    v8f acc0 = {}, acc1 = {};
    for (int kc = 0; kc < nChunk; ++kc) {
        const int kcBase = kc << 5;

        // B fragment: lane covers one tap, 16 consecutive channels.
        const int Kb  = kcBase + (laneHi << 4);
        const int tap = Kb >> cinShift;
        const int cb  = Kb & (Cin - 1);
        const int tyv = tap / TX;
        const int txv = tap - tyv * TX;
        const half_t* pB = ldsA + (tyv * LW + xi0 + txv) * CinP + cb;
        v16h bf = cat8(*(const v8h*)pB, *(const v8h*)(pB + 8));

        // A fragments: one contiguous 16-run in K per row (matches B).
        const int Ka = kcBase + (laneHi << 4);
        v16h a0 = cat8(*(const v8h*)(wRow0 + Ka), *(const v8h*)(wRow0 + Ka + 8));
        v16h a1 = cat8(*(const v8h*)(wRow1 + Ka), *(const v8h*)(wRow1 + Ka + 8));

        acc0 = __builtin_amdgcn_wmma_f32_16x16x32_f16(false, a0, false, bf,
                                                      (short)0, acc0, false, false);
        acc1 = __builtin_amdgcn_wmma_f32_16x16x32_f16(false, a1, false, bf,
                                                      (short)0, acc1, false, false);
    }

    // -------- store: lane L row r -> M = r + laneHi*8, N = lane&15 --------
    const int xOut = x0 + xi0;
#pragma unroll
    for (int r = 0; r < 8; ++r) {
        int m = r + laneHi * 8;
        float bv0 = bias ? bias[m] : 0.f;
        out[(((size_t)b * coutTotal + coutOffset + m) * H + y) * (size_t)W + xOut] =
            acc0[r] + bv0;
        int m1 = m + 16;
        if (m1 < Cout) {
            float bv1 = bias ? bias[m1] : 0.f;
            out[(((size_t)b * coutTotal + coutOffset + m1) * H + y) * (size_t)W + xOut] =
                acc1[r] + bv1;
        }
    }
}

// ---------------------------------------------------------------------------
// Modulated deformable 3x3 conv, Cout = 32.
//   v[c,t,pos] = mask_t * bilinear( relu(bn(src[c])) at (y+ky+dy, x+kx+dx) )
//   out[m,pos] = sum_{t,c} wd[m,c,t] * v[c,t,pos]   (K = t*Cin + c, WMMA)
// One tap per lane per K-chunk -> offset/mask/corner math amortized over 16
// channels; weights staged in LDS [m][K] and read as b128 fragments.
// ---------------------------------------------------------------------------
__global__ __launch_bounds__(BLOCK_T) void deform_conv_wmma(
    const float* __restrict__ src, const float* __restrict__ st,
    const float* __restrict__ gamma, const float* __restrict__ beta,
    const float* __restrict__ om,  const float* __restrict__ wd,
    float* __restrict__ out,
    int Cin, int cinShift, int coutTotal, int coutOffset, int H, int W)
{
    extern __shared__ __align__(16) char smem_raw[];
    const int Ktot  = Cin * 9;
    const int KtotP = Ktot + PADH;
    half_t* ldsW = (half_t*)smem_raw;             // [32][KtotP]

    const int b = blockIdx.z, y = blockIdx.y, x0 = blockIdx.x * XT;
    const int tid = threadIdx.x, lane = tid & 31, wid = tid >> 5;

    for (int idx = tid; idx < 32 * Ktot; idx += BLOCK_T) {
        int m = idx / Ktot;
        int K = idx - m * Ktot;
        int t = K >> cinShift;
        int c = K & (Cin - 1);
        ldsW[m * KtotP + K] = (half_t)wd[(m * Cin + c) * 9 + t];
    }
    __syncthreads();

    const int laneHi = (lane >= 16) ? 1 : 0;
    const int nIdx   = lane & 15;
    const int x      = x0 + wid * 16 + nIdx;
    const int mRow   = lane & 15;
    const float* mean = st;
    const float* rstd = st + Cin;
    const size_t plane = (size_t)H * W;
    const float* omB = om + (size_t)b * 27 * plane + (size_t)y * W + x;
    const half_t* wRow0 = ldsW + mRow * KtotP;
    const half_t* wRow1 = ldsW + (mRow + 16) * KtotP;

    v8f acc0 = {}, acc1 = {};
    const int nChunk = Ktot >> 5;
    for (int kc = 0; kc < nChunk; ++kc) {
        const int Kb    = (kc << 5) + (laneHi << 4);
        const int t     = Kb >> cinShift;      // tap: constant per lane per chunk
        const int cbase = Kb & (Cin - 1);

        float dy = omB[(size_t)(2 * t) * plane];
        float dx = omB[(size_t)(2 * t + 1) * plane];
        float mk = omB[(size_t)(18 + t) * plane];
        mk = 1.f / (1.f + __expf(-mk));        // sigmoid modulation

        float py = (float)y + (float)(t / 3 - 1) + dy;
        float px = (float)x + (float)(t % 3 - 1) + dx;
        float fy = floorf(py), fx = floorf(px);
        float wy = py - fy,    wx = px - fx;
        int y0i = (int)fy, x0i = (int)fx;
        int y1i = y0i + 1, x1i = x0i + 1;
        bool vy0 = (y0i >= 0) & (y0i < H), vy1 = (y1i >= 0) & (y1i < H);
        bool vx0 = (x0i >= 0) & (x0i < W), vx1 = (x1i >= 0) & (x1i < W);
        float w00 = (vy0 && vx0) ? (1.f - wy) * (1.f - wx) * mk : 0.f;
        float w01 = (vy0 && vx1) ? (1.f - wy) * wx * mk : 0.f;
        float w10 = (vy1 && vx0) ? wy * (1.f - wx) * mk : 0.f;
        float w11 = (vy1 && vx1) ? wy * wx * mk : 0.f;
        int yc0 = iclamp(y0i, 0, H - 1), yc1 = iclamp(y1i, 0, H - 1);
        int xc0 = iclamp(x0i, 0, W - 1), xc1 = iclamp(x1i, 0, W - 1);

        v16h bf;
#pragma unroll
        for (int j = 0; j < 16; ++j) {
            int c = cbase + j;
            const float* p = src + ((size_t)b * Cin + c) * plane;
            float g  = gamma[c] * rstd[c];
            float bb = beta[c] - mean[c] * g;
            float s00 = fmaxf(p[(size_t)yc0 * W + xc0] * g + bb, 0.f);
            float s01 = fmaxf(p[(size_t)yc0 * W + xc1] * g + bb, 0.f);
            float s10 = fmaxf(p[(size_t)yc1 * W + xc0] * g + bb, 0.f);
            float s11 = fmaxf(p[(size_t)yc1 * W + xc1] * g + bb, 0.f);
            bf[j] = (half_t)(w00 * s00 + w01 * s01 + w10 * s10 + w11 * s11);
        }

        const int Ka = (kc << 5) + (laneHi << 4);
        v16h a0 = cat8(*(const v8h*)(wRow0 + Ka), *(const v8h*)(wRow0 + Ka + 8));
        v16h a1 = cat8(*(const v8h*)(wRow1 + Ka), *(const v8h*)(wRow1 + Ka + 8));

        acc0 = __builtin_amdgcn_wmma_f32_16x16x32_f16(false, a0, false, bf,
                                                      (short)0, acc0, false, false);
        acc1 = __builtin_amdgcn_wmma_f32_16x16x32_f16(false, a1, false, bf,
                                                      (short)0, acc1, false, false);
    }

#pragma unroll
    for (int r = 0; r < 8; ++r) {
        int m = r + laneHi * 8;
        out[(((size_t)b * coutTotal + coutOffset + m) * H + y) * (size_t)W + x] = acc0[r];
        out[(((size_t)b * coutTotal + coutOffset + m + 16) * H + y) * (size_t)W + x] = acc1[r];
    }
}

// ---------------------------------------------------------------------------
// Host-side pipeline
// ---------------------------------------------------------------------------
static inline size_t conv_smem(int TY, int TX, int Cin) {
    int LW = XT + TX - 1;
    return (size_t)(TY * LW * (Cin + PADH) + 32 * (Cin * TY * TX + PADH)) *
           sizeof(half_t);
}

extern "C" void kernel_launch(void* const* d_in, const int* in_sizes, int n_in,
                              void* d_out, int out_size, void* d_ws, size_t ws_size,
                              hipStream_t stream) {
    (void)in_sizes; (void)n_in; (void)out_size; (void)ws_size;
    const int Bn = 2, H = 224, W = 224;
    const size_t HW = (size_t)H * W;

    const float* x     = (const float*)d_in[0];
    const float* g12   = (const float*)d_in[1];
    const float* b12   = (const float*)d_in[2];
    const float* w12   = (const float*)d_in[3];
    const float* woff1 = (const float*)d_in[4];
    const float* boff1 = (const float*)d_in[5];
    const float* g13   = (const float*)d_in[6];
    const float* b13   = (const float*)d_in[7];
    const float* wd1   = (const float*)d_in[8];
    const float* gc    = (const float*)d_in[42];
    const float* bc    = (const float*)d_in[43];
    const float* wc    = (const float*)d_in[44];
    const float* gy    = (const float*)d_in[45];
    const float* by    = (const float*)d_in[46];
    const float* wyg   = (const float*)d_in[47];

    // workspace carve-up
    float* cur = (float*)d_ws;
    float* st_x   = cur; cur += 64;
    float* st_z   = cur; cur += 128;
    float* st_cat = cur; cur += 256;
    float* st_c2  = cur; cur += 64;
    float* buf_z   = cur; cur += (size_t)Bn * 64  * HW;   // 1x1/concat conv outputs
    float* buf_om  = cur; cur += (size_t)Bn * 27  * HW;   // offset+mask maps
    float* buf_cat = cur; cur += (size_t)Bn * 128 * HW;   // concat of deform outputs
    float* buf_c2  = cur; cur += (size_t)Bn * 32  * HW;   // penultimate

    dim3 cgrid(W / XT, H, Bn);
    dim3 cblk(BLOCK_T);

    // stats(x): shared by every branch's input BN
    channel_stats<<<32, 256, 0, stream>>>(x, 32, (int)HW, Bn, st_x);

    // ---------------- branch 1 (1x1 path) ----------------
    bnrelu_conv_wmma<1, 1><<<cgrid, cblk, conv_smem(1, 1, 32), stream>>>(
        x, st_x, g12, b12, w12, nullptr, buf_z,
        32, 5, 32, 32, 0, 0, 0, H, W);
    channel_stats<<<32, 256, 0, stream>>>(buf_z, 32, (int)HW, Bn, st_z);
    bnrelu_conv_wmma<3, 3><<<cgrid, cblk, conv_smem(3, 3, 32), stream>>>(
        buf_z, st_z, g13, b13, woff1, boff1, buf_om,
        32, 5, 27, 27, 0, 1, 1, H, W);
    deform_conv_wmma<<<cgrid, cblk, (size_t)32 * (32 * 9 + PADH) * 2, stream>>>(
        buf_z, st_z, g13, b13, buf_om, wd1, buf_cat,
        32, 5, 128, 0, H, W);

    // ---------------- branches 2..4 (1xk + kx1 factorized paths) ----------
#define RUN_BRANCH(KS, BASE, CATOFF)                                           \
    {                                                                          \
        const float* g2   = (const float*)d_in[(BASE) + 0];                    \
        const float* b2   = (const float*)d_in[(BASE) + 1];                    \
        const float* w2   = (const float*)d_in[(BASE) + 2];                    \
        const float* g3   = (const float*)d_in[(BASE) + 3];                    \
        const float* b3   = (const float*)d_in[(BASE) + 4];                    \
        const float* w3   = (const float*)d_in[(BASE) + 5];                    \
        const float* g4   = (const float*)d_in[(BASE) + 6];                    \
        const float* b4   = (const float*)d_in[(BASE) + 7];                    \
        const float* woff = (const float*)d_in[(BASE) + 8];                    \
        const float* boff = (const float*)d_in[(BASE) + 9];                    \
        const float* wd   = (const float*)d_in[(BASE) + 10];                   \
        const int ph = ((KS) - 1) / 2;                                         \
        bnrelu_conv_wmma<1, KS><<<cgrid, cblk, conv_smem(1, KS, 32),           \
                                  stream>>>(                                   \
            x, st_x, g2, b2, w2, nullptr, buf_z,                               \
            32, 5, 32, 64, 0, 0, ph, H, W);                                    \
        bnrelu_conv_wmma<KS, 1><<<cgrid, cblk, conv_smem(KS, 1, 32),           \
                                  stream>>>(                                   \
            x, st_x, g3, b3, w3, nullptr, buf_z,                               \
            32, 5, 32, 64, 32, ph, 0, H, W);                                   \
        channel_stats<<<64, 256, 0, stream>>>(buf_z, 64, (int)HW, Bn, st_z);   \
        bnrelu_conv_wmma<3, 3><<<cgrid, cblk, conv_smem(3, 3, 64), stream>>>(  \
            buf_z, st_z, g4, b4, woff, boff, buf_om,                           \
            64, 6, 27, 27, 0, 1, 1, H, W);                                     \
        deform_conv_wmma<<<cgrid, cblk, (size_t)32 * (64 * 9 + PADH) * 2,      \
                           stream>>>(                                          \
            buf_z, st_z, g4, b4, buf_om, wd, buf_cat,                          \
            64, 6, 128, (CATOFF), H, W);                                       \
    }

    RUN_BRANCH(3,  9, 32)
    RUN_BRANCH(5, 20, 64)
    RUN_BRANCH(7, 31, 96)
#undef RUN_BRANCH

    // ---------------- tail: concat -> 1x1 -> 3x3 ----------------
    channel_stats<<<128, 256, 0, stream>>>(buf_cat, 128, (int)HW, Bn, st_cat);
    bnrelu_conv_wmma<1, 1><<<cgrid, cblk, conv_smem(1, 1, 128), stream>>>(
        buf_cat, st_cat, gc, bc, wc, nullptr, buf_c2,
        128, 7, 32, 32, 0, 0, 0, H, W);
    channel_stats<<<32, 256, 0, stream>>>(buf_c2, 32, (int)HW, Bn, st_c2);
    bnrelu_conv_wmma<3, 3><<<cgrid, cblk, conv_smem(3, 3, 32), stream>>>(
        buf_c2, st_c2, gy, by, wyg, nullptr, (float*)d_out,
        32, 5, 32, 32, 0, 1, 1, H, W);
}
